// TernaryLinear_78589311582693
// MI455X (gfx1250) — matrix operand order repaired and checked by reading, not compiled, not run on hardware
//
#include <hip/hip_runtime.h>

// ---------------------------------------------------------------------------
// TernaryLinear + LoRA fused as:  out = x @ Weff^T + bias
//   Weff[o,k] = scale[o]*Wq[o,k] + sum_r lora_A[o,r]*lora_B[r,k]   (f16, in d_ws)
// Main GEMM: v_wmma_f32_16x16x32_f16, 64x64 wave tiles, async B->LDS DMA.
// ---------------------------------------------------------------------------

typedef _Float16 v8h  __attribute__((ext_vector_type(8)));
typedef _Float16 v16h __attribute__((ext_vector_type(16)));
typedef float    v8f  __attribute__((ext_vector_type(8)));
typedef float    v4f  __attribute__((ext_vector_type(4)));
typedef unsigned int v4u __attribute__((ext_vector_type(4)));
typedef int      v4i  __attribute__((ext_vector_type(4)));

typedef __attribute__((address_space(1))) v4i* glb_v4i_ptr;   // "__device__" AS
typedef __attribute__((address_space(3))) v4i* lds_v4i_ptr;   // LDS AS

#define IN_F   3072
#define OUT_F  3072
#define M_TOT  8192        // B*S = 2*4096
#define RANK   64

#define BM 128             // block M tile
#define BN 256             // block N tile
#define BK 32              // K step = one WMMA K
#define LDSW 40            // halves per LDS row: 32 data + 8 pad (80B, 16B-aligned)

#if __has_builtin(__builtin_amdgcn_global_load_async_to_lds_b128)
#define HAVE_ASYNC_LDS 1
#else
#define HAVE_ASYNC_LDS 0
#endif

__device__ __forceinline__ void wait_async_lds() {
#if HAVE_ASYNC_LDS
#if __has_builtin(__builtin_amdgcn_s_wait_asynccnt)
    __builtin_amdgcn_s_wait_asynccnt(0);
#else
    asm volatile("s_wait_asynccnt 0x0" ::: "memory");
#endif
#endif
}

// ---------------------------------------------------------------------------
// Kernel 1: Weff (f16) = scale*Wq + lora_A @ lora_B
// grid = (IN_F/256, OUT_F), block = 256
// ---------------------------------------------------------------------------
__global__ __launch_bounds__(256)
void weff_build_kernel(const int* __restrict__ wq,
                       const float* __restrict__ scale,
                       const float* __restrict__ lA,
                       const float* __restrict__ lB,
                       _Float16* __restrict__ weff)
{
    __shared__ float sA[RANK];
    const int o = blockIdx.y;
    const int k = blockIdx.x * 256 + threadIdx.x;
    if (threadIdx.x < RANK)
        sA[threadIdx.x] = lA[o * RANK + threadIdx.x];
    __syncthreads();

    float acc = scale[o] * (float)wq[(size_t)o * IN_F + k];
#pragma unroll 8
    for (int r = 0; r < RANK; ++r)
        acc += sA[r] * lB[r * IN_F + k];
    weff[(size_t)o * IN_F + k] = (_Float16)acc;
}

// ---------------------------------------------------------------------------
// Kernel 2: out[m, o] = sum_k x[m,k] * weff[o,k] + bias[o]
// grid = (OUT_F/BN, M_TOT/BM), block = 256 (8 waves: 2 in M x 4 in N)
// Wave tile 64x64 = 4x4 WMMA accumulators.
// ---------------------------------------------------------------------------
__global__ __launch_bounds__(256)
void ternary_lora_gemm(const float* __restrict__ x,
                       const _Float16* __restrict__ weff,
                       const float* __restrict__ bias,
                       float* __restrict__ out)
{
    __shared__ _Float16 As[2][BM * LDSW];   // 20 KB
    __shared__ _Float16 Bs[2][BN * LDSW];   // 40 KB

    const int t     = threadIdx.x;
    const int lane  = t & 31;
    const int wid   = t >> 5;          // 0..7
    const int wm    = wid & 1;         // wave M offset = wm*64
    const int wn    = wid >> 1;        // wave N offset = wn*64
    const int lrow  = lane & 15;
    const int khalf = lane >> 4;

    const int n0 = blockIdx.x * BN;
    const int m0 = blockIdx.y * BM;

    // A staging: each thread converts 16 f32 of a 128x32 tile (row = t>>1).
    const int arow  = t >> 1;
    const int ahalf = (t & 1) * 16;
    const float* xg = x + (size_t)(m0 + arow) * IN_F + ahalf;

    // B staging: thread t owns row t of the 256x32 tile (64B per K-step).
    const _Float16* wg = weff + (size_t)(n0 + t) * IN_F;

    v8f acc[4][4];
#pragma unroll
    for (int i = 0; i < 4; ++i)
#pragma unroll
        for (int j = 0; j < 4; ++j)
            acc[i][j] = (v8f){0.f, 0.f, 0.f, 0.f, 0.f, 0.f, 0.f, 0.f};

    v4f apf[4];
#if !HAVE_ASYNC_LDS
    v4u bpf[4];
#endif

    // ---------------- helpers (expanded inline) ----------------
#define LOAD_A_REGS(KT)                                                      \
    {                                                                        \
        const float* p = xg + (KT) * BK;                                     \
        _Pragma("unroll")                                                    \
        for (int q = 0; q < 4; ++q) apf[q] = *(const v4f*)(p + q * 4);       \
    }

#define STORE_A_LDS(BUF)                                                     \
    {                                                                        \
        _Float16* as = &As[(BUF)][arow * LDSW + ahalf];                      \
        _Pragma("unroll")                                                    \
        for (int q = 0; q < 2; ++q) {                                        \
            v8h h;                                                           \
            _Pragma("unroll")                                                \
            for (int e = 0; e < 4; ++e) {                                    \
                h[e]     = (_Float16)apf[2 * q][e];                          \
                h[4 + e] = (_Float16)apf[2 * q + 1][e];                      \
            }                                                                \
            *(v8h*)(as + q * 8) = h;                                         \
        }                                                                    \
    }

#if HAVE_ASYNC_LDS
#define ISSUE_B(KT, BUF)                                                     \
    {                                                                        \
        glb_v4i_ptr g = (glb_v4i_ptr)(wg + (KT) * BK);                       \
        lds_v4i_ptr l = (lds_v4i_ptr)&Bs[(BUF)][t * LDSW];                   \
        _Pragma("unroll")                                                    \
        for (int q = 0; q < 4; ++q)                                          \
            __builtin_amdgcn_global_load_async_to_lds_b128(g + q, l + q,     \
                                                           0, 0);            \
    }
#define COMMIT_B(BUF) wait_async_lds();
#else
#define ISSUE_B(KT, BUF)                                                     \
    {                                                                        \
        const _Float16* p = wg + (KT) * BK;                                  \
        _Pragma("unroll")                                                    \
        for (int q = 0; q < 4; ++q) bpf[q] = *(const v4u*)(p + q * 8);       \
    }
#define COMMIT_B(BUF)                                                        \
    {                                                                        \
        _Float16* bs = &Bs[(BUF)][t * LDSW];                                 \
        _Pragma("unroll")                                                    \
        for (int q = 0; q < 4; ++q) *(v4u*)(bs + q * 8) = bpf[q];            \
    }
#endif
    // ------------------------------------------------------------

    // Prologue: stage K-tile 0 into buffer 0.
    ISSUE_B(0, 0)
    LOAD_A_REGS(0)
    STORE_A_LDS(0)
    COMMIT_B(0)
    __syncthreads();

    const int NK = IN_F / BK;   // 96
    for (int kt = 0; kt < NK; ++kt) {
        const int  buf  = kt & 1;
        const bool more = (kt + 1) < NK;

        // Start next tile's data movement early (overlaps the WMMAs).
        if (more) {
            ISSUE_B(kt + 1, buf ^ 1)
            LOAD_A_REGS(kt + 1)
        }

        // B fragments (32x16 f16): lane(n=lrow) holds K[khalf*16 .. +15]
        // B[k][n] = weff[n][k] -> one contiguous 32B LDS read per fragment.
        v16h bf[4];
#pragma unroll
        for (int j = 0; j < 4; ++j) {
            const _Float16* p =
                &Bs[buf][(wn * 64 + j * 16 + lrow) * LDSW + khalf * 16];
            v8h lo = *(const v8h*)p;
            v8h hi = *(const v8h*)(p + 8);
            bf[j] = __builtin_shufflevector(lo, hi,
                0,1,2,3,4,5,6,7,8,9,10,11,12,13,14,15);
        }

        // A fragments loaded per-row inside the i-loop (shorter live range):
    // lane(m=lrow) holds K[khalf*16 .. +15].
#pragma unroll
        for (int i = 0; i < 4; ++i) {
            const _Float16* p =
                &As[buf][(wm * 64 + i * 16 + lrow) * LDSW + khalf * 16];
            v8h lo = *(const v8h*)p;
            v8h hi = *(const v8h*)(p + 8);
            v16h af = __builtin_shufflevector(lo, hi,
                0,1,2,3,4,5,6,7,8,9,10,11,12,13,14,15);
#pragma unroll
            for (int j = 0; j < 4; ++j)
                acc[i][j] = __builtin_amdgcn_wmma_f32_16x16x32_f16(
                    false, af, false, bf[j],
                    (short)0, acc[i][j], false, false);
        }

        if (more) {
            STORE_A_LDS(buf ^ 1)
            COMMIT_B(buf ^ 1)
        }
        __syncthreads();
    }

    // ---- epilogue: +bias, non-temporal f32 stores (streaming output) ----
    float bv[4];
#pragma unroll
    for (int j = 0; j < 4; ++j)
        bv[j] = bias[n0 + wn * 64 + j * 16 + lrow];

#pragma unroll
    for (int i = 0; i < 4; ++i) {
#pragma unroll
        for (int j = 0; j < 4; ++j) {
            const int c = n0 + wn * 64 + j * 16 + lrow;
#pragma unroll
            for (int r = 0; r < 8; ++r) {
                const int m = m0 + wm * 64 + i * 16 + r + 8 * khalf;
                __builtin_nontemporal_store(acc[i][j][r] + bv[j],
                                            &out[(size_t)m * OUT_F + c]);
            }
        }
    }
}

// ---------------------------------------------------------------------------
// Launch: inputs in setup_inputs() order: x, weight_q, scale, lora_A, lora_B, bias
// ---------------------------------------------------------------------------
extern "C" void kernel_launch(void* const* d_in, const int* in_sizes, int n_in,
                              void* d_out, int out_size, void* d_ws, size_t ws_size,
                              hipStream_t stream)
{
    const float* x     = (const float*)d_in[0];
    const int*   wq    = (const int*)d_in[1];
    const float* scale = (const float*)d_in[2];
    const float* lA    = (const float*)d_in[3];
    const float* lB    = (const float*)d_in[4];
    const float* bias  = (const float*)d_in[5];
    float*       out   = (float*)d_out;
    _Float16*    weff  = (_Float16*)d_ws;   // OUT_F*IN_F f16 = ~18.9 MB scratch

    dim3 g1(IN_F / 256, OUT_F);
    weff_build_kernel<<<g1, 256, 0, stream>>>(wq, scale, lA, lB, weff);

    dim3 g2(OUT_F / BN, M_TOT / BM);   // 12 x 64
    ternary_lora_gemm<<<g2, 256, 0, stream>>>(x, weff, bias, out);
}
